// ChebConvWithSpatialAttention_62062277427614
// MI455X (gfx1250) — matrix operand order repaired and checked by reading, not compiled, not run on hardware
//
#include <hip/hip_runtime.h>

typedef __attribute__((ext_vector_type(16))) __bf16 v16bf;
typedef __attribute__((ext_vector_type(8)))  __bf16 v8bf;
typedef __attribute__((ext_vector_type(4)))  __bf16 v4bf;
typedef __attribute__((ext_vector_type(8)))  float  v8f;
typedef __attribute__((ext_vector_type(4)))  float  v4f;

namespace {
constexpr int B_ = 16, T_ = 12, N_ = 1024, FIN = 64, FOUT = 64, K_ = 3;
constexpr int MT   = 64;        // output rows (i) per workgroup
constexpr int KT   = 32;        // bf16 WMMA K depth per step
constexpr int NT   = N_ / KT;   // 32 j-steps
constexpr int LDA  = 40;        // sA row stride (bf16): 80 B, 16B-aligned frags
constexpr int LDXT = 40;        // sXT row stride (X^T: [f][j])
constexpr int LDTT = 72;        // sThT row stride (Theta^T: [o][f]), 144 B
constexpr int LDR  = 72;        // sR row stride (rhs: [i][f]), 144 B
}

// A fragment (16x32 bf16, MxK), row-major [M][K] source, K contiguous.
// lane m=row; element e -> K = 16*half + e (contiguous, matches frag_bT).
__device__ __forceinline__ v16bf frag_a(const __bf16* base, int ld, int row,
                                        int half, int k0) {
  const __bf16* p = base + row * ld + k0 + half * 16;
  v16bf a;
#pragma unroll
  for (int e = 0; e < 16; ++e) {
    int kl = e;
    a[e] = p[kl];
  }
  return a;
}

// B fragment (32x16 bf16, KxN) read from [N][K] storage (K contiguous):
// lane col = lane%16, K = k0 + 16*half + e -> 16 contiguous bf16 per lane.
__device__ __forceinline__ v16bf frag_bT(const __bf16* p) {
  v16bf bb;
#pragma unroll
  for (int e = 0; e < 16; ++e) bb[e] = p[e];
  return bb;
}

// Stage tiles with float4 global loads (32-bit offsets -> saddr form):
// sA[i][j] = attn*cheb (64x32), sXT[f][j] = x transposed (64x32).
__device__ __forceinline__ void fill_tiles(__bf16* sA, __bf16* sXT,
                                           const float* __restrict__ Ab,
                                           const float* __restrict__ Ck,
                                           const float* __restrict__ xbt,
                                           int i0, int j0, int tid) {
#pragma unroll
  for (int u = 0; u < 2; ++u) {
    int li = tid + u * 256;                    // 0..511 (float4 index)
    int r  = li >> 3, c4 = (li & 7) << 2;      // A tile: 64 rows x 8 float4
    int gi = (i0 + r) * N_ + j0 + c4;
    v4f av = *(const v4f*)(Ab + gi);
    v4f cv = *(const v4f*)(Ck + gi);
    v4bf pk;
#pragma unroll
    for (int e = 0; e < 4; ++e) pk[e] = (__bf16)(av[e] * cv[e]);
    *(v4bf*)&sA[r * LDA + c4] = pk;            // ds_store_b64

    int rx = li >> 4, cx4 = (li & 15) << 2;    // X tile: 32 rows x 16 float4
    v4f xv = *(const v4f*)(xbt + (j0 + rx) * FIN + cx4);
#pragma unroll
    for (int e = 0; e < 4; ++e)                // transpose -> scalar b16 stores
      sXT[(cx4 + e) * LDXT + rx] = (__bf16)xv[e];
  }
}

__global__ __launch_bounds__(256)
void cheb_attn_wmma_kernel(const float* __restrict__ x,
                           const float* __restrict__ attn,
                           const float* __restrict__ cheb,
                           const float* __restrict__ theta,
                           float* __restrict__ out) {
  __shared__ __align__(16) __bf16 sA[2][MT * LDA];        // A_k tiles (dbuf)
  __shared__ __align__(16) __bf16 sXT[2][FIN * LDXT];     // X^T tiles (dbuf)
  __shared__ __align__(16) __bf16 sThT[K_ * FOUT * LDTT]; // Theta^T [k][o][f]
  __shared__ __align__(16) __bf16 sR[MT * LDR];           // rhs [i][f]

  const int tid  = threadIdx.x;
  const int lane = tid & 31;
  const int half = lane >> 4;
  const int m16  = lane & 15;
  const int wave = tid >> 5;
  const int wr   = wave >> 1;   // 0..3 : 32-wide M group (f in s1, o in s2)
  const int wc   = wave & 1;    // 0..1 : 32-wide N group (i)

  const int i0 = blockIdx.x * MT;
  const int t  = blockIdx.y;
  const int b  = blockIdx.z;

  // Preload Theta transposed (f32 -> bf16): sThT[k][o][f] = Theta[k][f][o].
#pragma unroll 1
  for (int li = tid; li < K_ * FIN * FOUT; li += 256) {
    int k = li >> 12;
    int f = (li >> 6) & 63;
    int o = li & 63;
    sThT[(k * FOUT + o) * LDTT + f] = (__bf16)theta[li];
  }

  v8f oacc[2][2] = {};   // out^T tiles: M=o (wr), N=i (wc)
  const float* xbt = x + (size_t)(b * T_ + t) * N_ * FIN;
  const float* Ab  = attn + (size_t)b * N_ * N_;

  for (int k = 0; k < K_; ++k) {
    v8f racc[2][2] = {};   // rhs^T tiles: M=f (wr), N=i (wc)
    const float* Ck = cheb + (size_t)k * N_ * N_;

    // Prologue: stage first tile into buffer 0.
    fill_tiles(sA[0], sXT[0], Ab, Ck, xbt, i0, 0, tid);

#pragma unroll 2
    for (int jt = 0; jt < NT; ++jt) {
      const int cur = jt & 1;
      __syncthreads();   // buf[cur] ready; buf[cur^1] readers done

      if (jt + 1 < NT) {
        fill_tiles(sA[cur ^ 1], sXT[cur ^ 1], Ab, Ck, xbt,
                   i0, (jt + 1) * KT, tid);
        int gi = (i0 + (tid >> 5)) * N_ + ((jt + 2) * KT + (tid & 31));
        if (jt + 2 < NT) {                     // global_prefetch_b8
          __builtin_prefetch(&Ab[gi], 0, 1);
          __builtin_prefetch(&Ck[gi], 0, 1);
        }
      }

      // rhs^T = X^T @ A_k^T : A-frag rows f from sXT, B-frag cols i from sA.
      v16bf af[2], bf[2];
#pragma unroll
      for (int ft = 0; ft < 2; ++ft)
        af[ft] = frag_a(sXT[cur], LDXT, wr * 32 + ft * 16 + m16, half, 0);
#pragma unroll
      for (int it = 0; it < 2; ++it)
        bf[it] = frag_bT(&sA[cur][(wc * 32 + it * 16 + m16) * LDA + half * 16]);
#pragma unroll
      for (int ft = 0; ft < 2; ++ft)
#pragma unroll
        for (int it = 0; it < 2; ++it)
          racc[ft][it] = __builtin_amdgcn_wmma_f32_16x16x32_bf16(
              false, af[ft], false, bf[it], (short)0, racc[ft][it],
              false, false);
    }

    // Dump rhs^T accumulators to sR[i][f]: lane holds fixed i, contiguous f
    // -> one packed-bf16 ds_store_b128 per 16x16 tile. (No barrier needed
    // before: previous readers of sR are fenced by the j-loop barriers.)
#pragma unroll
    for (int ft = 0; ft < 2; ++ft)
#pragma unroll
      for (int it = 0; it < 2; ++it) {
        v8bf pk;
#pragma unroll
        for (int r = 0; r < 8; ++r) pk[r] = (__bf16)racc[ft][it][r];
        int i_row = wc * 32 + it * 16 + m16;
        int f_col = wr * 32 + ft * 16 + 8 * half;
        *(v8bf*)&sR[i_row * LDR + f_col] = pk;
      }
    __syncthreads();   // sR visible to all waves

    // Stage 2: out^T += Theta_k^T @ rhs^T (A from sThT, B from sR).
#pragma unroll
    for (int ks = 0; ks < 2; ++ks) {
      v16bf a2[2], b2[2];
#pragma unroll
      for (int ot = 0; ot < 2; ++ot)
        a2[ot] = frag_a(sThT + k * FOUT * LDTT, LDTT,
                        wr * 32 + ot * 16 + m16, half, ks * 32);
#pragma unroll
      for (int it = 0; it < 2; ++it)
        b2[it] = frag_bT(&sR[(wc * 32 + it * 16 + m16) * LDR +
                             ks * 32 + half * 16]);
#pragma unroll
      for (int ot = 0; ot < 2; ++ot)
#pragma unroll
        for (int it = 0; it < 2; ++it)
          oacc[ot][it] = __builtin_amdgcn_wmma_f32_16x16x32_bf16(
              false, a2[ot], false, b2[it], (short)0, oacc[ot][it],
              false, false);
    }
    // No trailing barrier: next k's first j-loop barrier fences sR/sA reuse.
  }

  // Epilogue: ReLU + vectorized f32 stores. oacc element (r, lane):
  // o = wr*32 + ot*16 + 8*half + r (contiguous), i = wc*32 + it*16 + m16.
  float* op = out + ((size_t)(b * T_ + t) * N_ + i0) * FOUT;
#pragma unroll
  for (int ot = 0; ot < 2; ++ot)
#pragma unroll
    for (int it = 0; it < 2; ++it) {
      int i_row = wc * 32 + it * 16 + m16;
      int o0    = wr * 32 + ot * 16 + 8 * half;
      v4f lo, hi;
#pragma unroll
      for (int r = 0; r < 4; ++r) {
        float vl = oacc[ot][it][r];
        float vh = oacc[ot][it][r + 4];
        lo[r] = vl > 0.f ? vl : 0.f;
        hi[r] = vh > 0.f ? vh : 0.f;
      }
      float* dst = op + (size_t)i_row * FOUT + o0;
      *(v4f*)(dst)     = lo;
      *(v4f*)(dst + 4) = hi;
    }
}

extern "C" void kernel_launch(void* const* d_in, const int* in_sizes, int n_in,
                              void* d_out, int out_size, void* d_ws,
                              size_t ws_size, hipStream_t stream) {
  (void)in_sizes; (void)n_in; (void)out_size; (void)d_ws; (void)ws_size;
  const float* x    = (const float*)d_in[0];  // [B,T,N,F_IN]
  const float* attn = (const float*)d_in[1];  // [B,N,N]
  const float* cheb = (const float*)d_in[2];  // [K,N,N]
  const float* th   = (const float*)d_in[3];  // [K,F_IN,F_OUT]
  float* out = (float*)d_out;                 // [B,T,N,F_OUT]

  dim3 grid(N_ / MT, T_, B_);   // (16, 12, 16)
  cheb_attn_wmma_kernel<<<grid, 256, 0, stream>>>(x, attn, cheb, th, out);
}
